// LlamaAttention_18803366822137
// MI455X (gfx1250) — matrix operand order repaired and checked by reading, not compiled, not run on hardware
//
#include <hip/hip_runtime.h>

// ---------------- problem constants ----------------
#define HIDDEN   4096
#define NHEADS   32
#define HDIM     128
#define SEQ      2048
#define BATCH    2
#define MTOT     (BATCH * SEQ)          // 4096 tokens
#define NQKV     (3 * HIDDEN)           // 12288
#define SCALING  0.08838834764831845f   // 128^-0.5

typedef __attribute__((ext_vector_type(16))) __bf16 bf16x16;
typedef __attribute__((ext_vector_type(8)))  __bf16 bf16x8;
typedef __attribute__((ext_vector_type(8)))  float  f32x8;
typedef __attribute__((ext_vector_type(4)))  float  f32x4;
typedef __attribute__((ext_vector_type(4)))  unsigned int u32x4;
typedef __attribute__((ext_vector_type(8)))  int    i32x8;
typedef __attribute__((ext_vector_type(4)))  int    i32x4;

static __device__ __forceinline__ f32x8 wmma_bf16(bf16x16 a, bf16x16 b, f32x8 c) {
  return __builtin_amdgcn_wmma_f32_16x16x32_bf16(false, a, false, b, (short)0, c,
                                                 false, false);
}

// A-fragment (CDNA5 16-bit A layout): two contiguous 16B halves, 16 elems apart.
static __device__ __forceinline__ bf16x16 make_afrag(const __bf16* p) {
  bf16x8 lo = *(const bf16x8*)(p);
  bf16x8 hi = *(const bf16x8*)(p + 16);
  return __builtin_shufflevector(lo, hi, 0,1,2,3,4,5,6,7,8,9,10,11,12,13,14,15);
}
// B-fragment: two contiguous 16B halves, 16 elems apart (same K order as A).
static __device__ __forceinline__ bf16x16 load_bfrag(const __bf16* p) {
  bf16x8 lo = *(const bf16x8*)(p);
  bf16x8 hi = *(const bf16x8*)(p + 16);
  return __builtin_shufflevector(lo, hi, 0,1,2,3,4,5,6,7,8,9,10,11,12,13,14,15);
}
static __device__ __forceinline__ void zero8(f32x8& v) {
  #pragma unroll
  for (int j = 0; j < 8; ++j) v[j] = 0.0f;
}

// =====================================================================
// Kernel 0: one-time f32 -> bf16 pack (makes weights L2-resident: 100MB < 192MB)
// =====================================================================
__global__ __launch_bounds__(256) void cvt_bf16_kernel(
    const float* __restrict__ src, __bf16* __restrict__ dst) {
  const size_t i = ((size_t)blockIdx.x * 256 + threadIdx.x) * 8;
  f32x4 a = *(const f32x4*)(src + i);
  f32x4 b = *(const f32x4*)(src + i + 4);
  bf16x8 o;
  o[0] = (__bf16)a.x; o[1] = (__bf16)a.y; o[2] = (__bf16)a.z; o[3] = (__bf16)a.w;
  o[4] = (__bf16)b.x; o[5] = (__bf16)b.y; o[6] = (__bf16)b.z; o[7] = (__bf16)b.w;
  *(bf16x8*)(dst + i) = o;
}

// =====================================================================
// Kernel 1: qkv = hs @ W_qkv (all-bf16 in), fused RoPE epilogue.
// M-tile 256 (8 waves x 32 rows), N-tile 128 (= one head), KT = 64.
// Register double-buffered global->LDS staging.
// grid = (MTOT/256, NQKV/128)
// =====================================================================
__global__ __launch_bounds__(256) void qkv_rope_kernel(
    const __bf16* __restrict__ hs, const __bf16* __restrict__ Wqkv,
    const int* __restrict__ positions,
    __bf16* __restrict__ qb, __bf16* __restrict__ kb, __bf16* __restrict__ vb) {
  __shared__ __align__(32) __bf16 Asm[256 * 64];   // [row][k]
  __shared__ __align__(32) __bf16 Bsm[128 * 64];   // [n][k] (transposed)

  const int tid = threadIdx.x, w = tid >> 5, lane = tid & 31;
  const int lm = lane & 15, hi = lane >> 4;
  const int m0 = blockIdx.x * 256;
  const int by = blockIdx.y;
  const size_t n0 = (size_t)by * 128;

  f32x8 acc[2][8];
  #pragma unroll
  for (int rs = 0; rs < 2; ++rs)
    #pragma unroll
    for (int t = 0; t < 8; ++t) zero8(acc[rs][t]);

  bf16x16 areg[4], breg[2];
  int an_r[4], an_c[4], bn_k[2], bn_n[2];
  #pragma unroll
  for (int p = 0; p < 4; ++p) {
    const int idx = tid * 16 + p * 4096;   // A tile: 256x64
    an_r[p] = idx >> 6; an_c[p] = idx & 63;
  }
  #pragma unroll
  for (int p = 0; p < 2; ++p) {
    const int idx = tid * 16 + p * 4096;   // B tile: 64x128
    bn_k[p] = idx >> 7; bn_n[p] = idx & 127;
  }

  auto load_tiles = [&](int k0) {
    #pragma unroll
    for (int p = 0; p < 4; ++p)
      areg[p] = *(const bf16x16*)(hs + (size_t)(m0 + an_r[p]) * HIDDEN + k0 + an_c[p]);
    #pragma unroll
    for (int p = 0; p < 2; ++p)
      breg[p] = *(const bf16x16*)(Wqkv + (size_t)(k0 + bn_k[p]) * NQKV + n0 + bn_n[p]);
  };
  auto store_tiles = [&]() {
    #pragma unroll
    for (int p = 0; p < 4; ++p)
      *(bf16x16*)(Asm + an_r[p] * 64 + an_c[p]) = areg[p];
    #pragma unroll
    for (int p = 0; p < 2; ++p)
      #pragma unroll
      for (int j = 0; j < 16; ++j)
        Bsm[(bn_n[p] + j) * 64 + bn_k[p]] = breg[p][j];
  };

  load_tiles(0);
  for (int k0 = 0; k0 < HIDDEN; k0 += 64) {
    store_tiles();
    __syncthreads();
    if (k0 + 64 < HIDDEN) load_tiles(k0 + 64);   // overlaps with WMMA burst
    #pragma unroll
    for (int c = 0; c < 2; ++c) {
      bf16x16 a0 = make_afrag(Asm + (w * 32 + lm) * 64 + c * 32 + hi * 8);
      bf16x16 a1 = make_afrag(Asm + (w * 32 + 16 + lm) * 64 + c * 32 + hi * 8);
      bf16x16 bf[4];
      #pragma unroll
      for (int g = 0; g < 2; ++g) {
        #pragma unroll
        for (int t = 0; t < 4; ++t)
          bf[t] = load_bfrag(Bsm + ((g * 4 + t) * 16 + lm) * 64 + c * 32 + hi * 8);
        #pragma unroll
        for (int t = 0; t < 4; ++t) {          // each B-frag feeds two WMMAs
          acc[0][g * 4 + t] = wmma_bf16(a0, bf[t], acc[0][g * 4 + t]);
          acc[1][g * 4 + t] = wmma_bf16(a1, bf[t], acc[1][g * 4 + t]);
        }
      }
    }
    __syncthreads();
  }

  // ---- epilogue: scatter to [B,H,S,D] bf16, RoPE on q/k ----
  const int which = by >> 5;  // 0=q 1=k 2=v
  const int h     = by & 31;
  #pragma unroll
  for (int rs = 0; rs < 2; ++rs) {
    if (which == 2) {
      #pragma unroll
      for (int t = 0; t < 8; ++t)
        #pragma unroll
        for (int i = 0; i < 8; ++i) {
          const int m = m0 + w * 32 + rs * 16 + i + hi * 8;
          const size_t dst =
              ((((size_t)(m >> 11) * NHEADS + h) * SEQ) + (m & (SEQ - 1))) * HDIM +
              t * 16 + lm;
          vb[dst] = (__bf16)acc[rs][t][i];
        }
    } else {
      __bf16* out = (which == 0) ? qb : kb;
      #pragma unroll
      for (int i = 0; i < 8; ++i) {
        const int m = m0 + w * 32 + rs * 16 + i + hi * 8;
        const float pos = (float)positions[m];
        const size_t base =
            ((((size_t)(m >> 11) * NHEADS + h) * SEQ) + (m & (SEQ - 1))) * (size_t)HDIM;
        #pragma unroll
        for (int t = 0; t < 4; ++t) {
          const int dm = t * 16 + lm;          // 0..63
          const float freq = __powf(10000.0f, -(float)dm * (1.0f / 64.0f));
          const float ang = pos * freq;
          const float cs = __cosf(ang), sn = __sinf(ang);
          const float x1 = acc[rs][t][i], x2 = acc[rs][t + 4][i];  // (d, d+64) lane-local
          out[base + dm]      = (__bf16)(x1 * cs - x2 * sn);
          out[base + dm + 64] = (__bf16)(x2 * cs + x1 * sn);
        }
      }
    }
  }
}

// =====================================================================
// Kernel 2: flash attention (causal, online softmax), all-WMMA.
// K tile staged by the Tensor Data Mover (TENSOR_LOAD_TO_LDS, TENSORcnt);
// V tile staged manually (TDM cannot transpose) concurrently with the DMA.
// grid = (SEQ/128, NHEADS, BATCH); each wave owns 16 q rows
// =====================================================================
__global__ __launch_bounds__(256) void attn_kernel(
    const __bf16* __restrict__ qb, const __bf16* __restrict__ kbuf,
    const __bf16* __restrict__ vb, __bf16* __restrict__ attn) {
  __shared__ __align__(32) __bf16 Ksm[32 * 128];   // [key][d] row-major (TDM dest)
  __shared__ __align__(32) __bf16 Vsm[128 * 32];   // [d][kv]  transposed
  __shared__ __align__(32) __bf16 Psm[8 * 16 * 32];// per-wave P re-shape pad

  const int tid = threadIdx.x, w = tid >> 5, lane = tid & 31;
  const int lm = lane & 15, hi = lane >> 4;
  const int q0 = blockIdx.x * 128;
  const int h = blockIdx.y, b = blockIdx.z;
  const size_t headoff = (((size_t)b * NHEADS + h) * SEQ) * HDIM;
  const __bf16* qh = qb + headoff;
  const __bf16* kh = kbuf + headoff;
  const __bf16* vh = vb + headoff;

  // ---- Tensor DMA descriptor (constant part), per CDNA5 ISA ----
  // group1: data_size=2B, tensor_dim0=128, tensor_dim1=2048, tile=128x32,
  //         tensor_dim0_stride=128 (fields straddle 32-bit words)
  const unsigned lds_k = (unsigned)(size_t)(void*)Ksm;   // LDS byte address
  const i32x8 g1 = {0x00010000,          // data_size=1 (2 bytes) at bits[17:16]
                    (int)(128u << 16),   // tensor_dim0[15:0] in bits[63:48]
                    (int)(2048u << 16),  // tensor_dim1[15:0] in bits[95:80]
                    (int)(128u << 16),   // tile_dim0=128 in bits[127:112]
                    32,                  // tile_dim1=32 in bits[143:128]
                    128,                 // tensor_dim0_stride in bits[191:160]
                    0, 0};
  const i32x4 gz  = {0, 0, 0, 0};
  const i32x8 gz8 = {0, 0, 0, 0, 0, 0, 0, 0};

  bf16x16 aq[4];
  {
    const __bf16* qrow = qh + (size_t)(q0 + w * 16 + lm) * HDIM + hi * 8;
    #pragma unroll
    for (int c = 0; c < 4; ++c) aq[c] = make_afrag(qrow + c * 32);
  }

  f32x8 o[8];
  #pragma unroll
  for (int t = 0; t < 8; ++t) zero8(o[t]);
  float rmax[8], rsum[8];
  #pragma unroll
  for (int i = 0; i < 8; ++i) { rmax[i] = -3.0e38f; rsum[i] = 0.0f; }

  const int q_lo = q0 + w * 16;
  const int kv_end = q0 + 128;

  for (int kv0 = 0; kv0 < kv_end; kv0 += 32) {
    __syncthreads();
    if (w == 0) {   // one wave issues the K-tile DMA (descriptor is uniform)
      const unsigned long long ga =
          (unsigned long long)(size_t)(const void*)(kh + (size_t)kv0 * HDIM);
      u32x4 g0 = {1u,                                   // count=1, gather off
                  lds_k,                                // lds_addr bits[63:32]
                  (unsigned)ga,                         // global_addr[31:0]
                  (unsigned)(ga >> 32) | (2u << 30)};   // addr[56:32] | type=2
      __builtin_amdgcn_tensor_load_to_lds(g0, g1, gz, gz, gz8, 0);
    }
    { // all threads: manual transposed stage of V (overlaps with the TDM)
      const int idx = tid * 16;
      const int key = idx >> 7, d = idx & 127;
      const __bf16* vs = vh + (size_t)(kv0 + key) * HDIM + d;
      #pragma unroll
      for (int j = 0; j < 16; ++j) Vsm[(d + j) * 32 + key] = vs[j];
      if (kv0 + 32 < kv_end)
        __builtin_prefetch(vh + (size_t)(kv0 + 32 + key) * HDIM + d, 0, 1);
    }
    if (w == 0) __builtin_amdgcn_s_wait_tensorcnt((short)0);
    __syncthreads();   // publishes TDM-written Ksm + Vsm to all waves

    if (kv0 <= q_lo + 15) {
      // ---- scores: batch all 8 K-fragments, then back-to-back WMMAs ----
      bf16x16 bk[8];
      #pragma unroll
      for (int c = 0; c < 4; ++c)
        #pragma unroll
        for (int half = 0; half < 2; ++half)
          bk[c * 2 + half] =
              load_bfrag(Ksm + (half * 16 + lm) * 128 + c * 32 + hi * 8);
      f32x8 sc[2];
      zero8(sc[0]); zero8(sc[1]);
      #pragma unroll
      for (int c = 0; c < 4; ++c) {
        sc[0] = wmma_bf16(aq[c], bk[c * 2 + 0], sc[0]);
        sc[1] = wmma_bf16(aq[c], bk[c * 2 + 1], sc[1]);
      }
      // ---- scale + causal mask + tile row-max ----
      float tmax[8];
      #pragma unroll
      for (int i = 0; i < 8; ++i) tmax[i] = -3.0e38f;
      #pragma unroll
      for (int half = 0; half < 2; ++half)
        #pragma unroll
        for (int i = 0; i < 8; ++i) {
          const int r = q_lo + i + hi * 8;
          const int col = kv0 + half * 16 + lm;
          float s = sc[half][i] * SCALING;
          if (col > r) s = -3.0e38f;
          sc[half][i] = s;
          tmax[i] = fmaxf(tmax[i], s);
        }
      #pragma unroll
      for (int i = 0; i < 8; ++i)
        #pragma unroll
        for (int msk = 1; msk < 16; msk <<= 1)
          tmax[i] = fmaxf(tmax[i], __shfl_xor(tmax[i], msk, 32));
      // ---- online softmax update ----
      float corr[8], tsum[8];
      #pragma unroll
      for (int i = 0; i < 8; ++i) {
        const float nm = fmaxf(rmax[i], tmax[i]);
        corr[i] = __expf(rmax[i] - nm);
        rmax[i] = nm;
        tsum[i] = 0.0f;
      }
      #pragma unroll
      for (int half = 0; half < 2; ++half)
        #pragma unroll
        for (int i = 0; i < 8; ++i) {
          const float p = __expf(sc[half][i] - rmax[i]);
          sc[half][i] = p;
          tsum[i] += p;
        }
      #pragma unroll
      for (int i = 0; i < 8; ++i) {
        #pragma unroll
        for (int msk = 1; msk < 16; msk <<= 1)
          tsum[i] += __shfl_xor(tsum[i], msk, 32);
        rsum[i] = rsum[i] * corr[i] + tsum[i];
      }
      #pragma unroll
      for (int t = 0; t < 8; ++t)
        #pragma unroll
        for (int i = 0; i < 8; ++i) o[t][i] *= corr[i];
      // ---- re-shape P (C layout -> A layout) via per-wave LDS pad ----
      __bf16* Pl = Psm + w * 512;
      #pragma unroll
      for (int half = 0; half < 2; ++half)
        #pragma unroll
        for (int i = 0; i < 8; ++i)
          Pl[(i + hi * 8) * 32 + half * 16 + lm] = (__bf16)sc[half][i];
      bf16x16 pa = make_afrag(Pl + lm * 32 + hi * 8);  // same-wave DS is ordered
      // ---- O += P V: batch V-fragments, back-to-back WMMAs ----
      bf16x16 bv[8];
      #pragma unroll
      for (int t = 0; t < 8; ++t)
        bv[t] = load_bfrag(Vsm + (t * 16 + lm) * 32 + hi * 8);
      #pragma unroll
      for (int t = 0; t < 8; ++t) o[t] = wmma_bf16(pa, bv[t], o[t]);
    }
  }

  // ---- normalize and store attn in [token][HIDDEN] bf16 ----
  float inv[8];
  #pragma unroll
  for (int i = 0; i < 8; ++i) inv[i] = 1.0f / rsum[i];
  #pragma unroll
  for (int t = 0; t < 8; ++t)
    #pragma unroll
    for (int i = 0; i < 8; ++i) {
      const int r = q_lo + i + hi * 8;
      const size_t dst = (size_t)(b * SEQ + r) * HIDDEN + h * HDIM + t * 16 + lm;
      attn[dst] = (__bf16)(o[t][i] * inv[i]);
    }
}

// =====================================================================
// Kernel 3: out = attn(bf16) @ W_o(bf16) -> f32. Same tiling as K1.
// grid = (MTOT/256, HIDDEN/128)
// =====================================================================
__global__ __launch_bounds__(256) void out_gemm_kernel(
    const __bf16* __restrict__ attn, const __bf16* __restrict__ Wo,
    float* __restrict__ out) {
  __shared__ __align__(32) __bf16 Asm[256 * 64];
  __shared__ __align__(32) __bf16 Bsm[128 * 64];

  const int tid = threadIdx.x, w = tid >> 5, lane = tid & 31;
  const int lm = lane & 15, hi = lane >> 4;
  const int m0 = blockIdx.x * 256;
  const size_t n0 = (size_t)blockIdx.y * 128;

  f32x8 acc[2][8];
  #pragma unroll
  for (int rs = 0; rs < 2; ++rs)
    #pragma unroll
    for (int t = 0; t < 8; ++t) zero8(acc[rs][t]);

  bf16x16 areg[4], breg[2];
  int an_r[4], an_c[4], bn_k[2], bn_n[2];
  #pragma unroll
  for (int p = 0; p < 4; ++p) {
    const int idx = tid * 16 + p * 4096;
    an_r[p] = idx >> 6; an_c[p] = idx & 63;
  }
  #pragma unroll
  for (int p = 0; p < 2; ++p) {
    const int idx = tid * 16 + p * 4096;
    bn_k[p] = idx >> 7; bn_n[p] = idx & 127;
  }

  auto load_tiles = [&](int k0) {
    #pragma unroll
    for (int p = 0; p < 4; ++p)
      areg[p] = *(const bf16x16*)(attn + (size_t)(m0 + an_r[p]) * HIDDEN + k0 + an_c[p]);
    #pragma unroll
    for (int p = 0; p < 2; ++p)
      breg[p] = *(const bf16x16*)(Wo + (size_t)(k0 + bn_k[p]) * HIDDEN + n0 + bn_n[p]);
  };
  auto store_tiles = [&]() {
    #pragma unroll
    for (int p = 0; p < 4; ++p)
      *(bf16x16*)(Asm + an_r[p] * 64 + an_c[p]) = areg[p];
    #pragma unroll
    for (int p = 0; p < 2; ++p)
      #pragma unroll
      for (int j = 0; j < 16; ++j)
        Bsm[(bn_n[p] + j) * 64 + bn_k[p]] = breg[p][j];
  };

  load_tiles(0);
  for (int k0 = 0; k0 < HIDDEN; k0 += 64) {
    store_tiles();
    __syncthreads();
    if (k0 + 64 < HIDDEN) load_tiles(k0 + 64);
    #pragma unroll
    for (int c = 0; c < 2; ++c) {
      bf16x16 a0 = make_afrag(Asm + (w * 32 + lm) * 64 + c * 32 + hi * 8);
      bf16x16 a1 = make_afrag(Asm + (w * 32 + 16 + lm) * 64 + c * 32 + hi * 8);
      bf16x16 bf[4];
      #pragma unroll
      for (int g = 0; g < 2; ++g) {
        #pragma unroll
        for (int t = 0; t < 4; ++t)
          bf[t] = load_bfrag(Bsm + ((g * 4 + t) * 16 + lm) * 64 + c * 32 + hi * 8);
        #pragma unroll
        for (int t = 0; t < 4; ++t) {
          acc[0][g * 4 + t] = wmma_bf16(a0, bf[t], acc[0][g * 4 + t]);
          acc[1][g * 4 + t] = wmma_bf16(a1, bf[t], acc[1][g * 4 + t]);
        }
      }
    }
    __syncthreads();
  }

  #pragma unroll
  for (int rs = 0; rs < 2; ++rs)
    #pragma unroll
    for (int t = 0; t < 8; ++t)
      #pragma unroll
      for (int i = 0; i < 8; ++i) {
        const int m = m0 + w * 32 + rs * 16 + i + hi * 8;
        out[(size_t)m * HIDDEN + n0 + t * 16 + lm] = acc[rs][t][i];
      }
}

// =====================================================================
extern "C" void kernel_launch(void* const* d_in, const int* in_sizes, int n_in,
                              void* d_out, int out_size, void* d_ws, size_t ws_size,
                              hipStream_t stream) {
  const int*   positions = (const int*)d_in[0];
  const float* hs_f      = (const float*)d_in[1];
  const float* Wqkv_f    = (const float*)d_in[2];
  const float* Wo_f      = (const float*)d_in[3];
  float*       out       = (float*)d_out;

  char* ws = (char*)d_ws;
  const size_t elems = (size_t)BATCH * NHEADS * SEQ * HDIM;     // 16.78M
  __bf16* qb      = (__bf16*)(ws);
  __bf16* kb      = (__bf16*)(ws + 2 * elems);
  __bf16* vb      = (__bf16*)(ws + 4 * elems);
  __bf16* attn    = (__bf16*)(ws + 6 * elems);
  __bf16* hs_b    = (__bf16*)(ws + 8 * elems);
  __bf16* wo_b    = (__bf16*)(ws + 10 * elems);
  __bf16* wqkv_b  = (__bf16*)(ws + 12 * elems);                 // 3*elems long

  const size_t n_hs   = (size_t)MTOT * HIDDEN;     // 16.78M
  const size_t n_wo   = (size_t)HIDDEN * HIDDEN;   // 16.78M
  const size_t n_wqkv = (size_t)HIDDEN * NQKV;     // 50.33M

  cvt_bf16_kernel<<<dim3(n_hs / (8 * 256)), 256, 0, stream>>>(hs_f, hs_b);
  cvt_bf16_kernel<<<dim3(n_wo / (8 * 256)), 256, 0, stream>>>(Wo_f, wo_b);
  cvt_bf16_kernel<<<dim3(n_wqkv / (8 * 256)), 256, 0, stream>>>(Wqkv_f, wqkv_b);

  qkv_rope_kernel<<<dim3(MTOT / 256, NQKV / 128), 256, 0, stream>>>(
      hs_b, wqkv_b, positions, qb, kb, vb);
  attn_kernel<<<dim3(SEQ / 128, NHEADS, BATCH), 256, 0, stream>>>(
      qb, kb, vb, attn);
  out_gemm_kernel<<<dim3(MTOT / 256, HIDDEN / 128), 256, 0, stream>>>(
      attn, wo_b, out);
}
